// HOG_47794396070684
// MI455X (gfx1250) — compile-verified, matrix operand order checked by reading, not run on hardware
//
#include <hip/hip_runtime.h>
#include <stdint.h>

#define NBINS 9
#define IMG 512
#define CROWS 64            // 512 / 8 cell rows/cols
#define CHUNK 20            // 16 data dwords + 4 pad dwords per lane-chunk
#define BANDSTRIDE 640      // 32 chunks * 20 dwords per image row
#define NCOLS 576           // 64 cells * 9 bins per cell-row

typedef _Float16 v16h __attribute__((ext_vector_type(16)));
typedef float    v8f  __attribute__((ext_vector_type(8)));
typedef unsigned int u32x4 __attribute__((ext_vector_type(4)));
typedef int i32x8 __attribute__((ext_vector_type(8)));
typedef int i32x4 __attribute__((ext_vector_type(4)));

#if defined(__AMDGCN__) && __has_builtin(__builtin_amdgcn_tensor_load_to_lds) && __has_builtin(__builtin_amdgcn_s_wait_tensorcnt)
#define HOG_TDM 1
#else
#define HOG_TDM 0
#endif

#if defined(__AMDGCN__)
#if HOG_TDM
#warning "HOG device pass: TDM tensor_load_to_lds path ENABLED"
#else
#warning "HOG device pass: TDM builtin unavailable, fallback copy path"
#endif
#endif

__global__ __launch_bounds__(256) void hog_cdna5_kernel(const float* __restrict__ x,
                                                        float* __restrict__ out) {
  __shared__ __align__(16) float band[10 * BANDSTRIDE];   // 25.6 KB staged input band
  __shared__ float part[8 * NCOLS];                       // 18 KB per-wave partial hists

  const int tid = threadIdx.x;
  const int w   = tid >> 5;     // wave id 0..7  -> pixel row within cell
  const int L   = tid & 31;     // lane id       -> column chunk (cells 2L, 2L+1)
  const int cy  = blockIdx.x;   // cell row 0..63
  const int img = blockIdx.y;   // image index

  const int r0     = cy * 8 - 1;                       // first band row (may be -1)
  const int gstart = (r0 < 0) ? 0 : r0;
  const int gend   = (cy * 8 + 9 > IMG) ? IMG : (cy * 8 + 9);
  const int nrows  = gend - gstart;                    // 9 or 10
  const int rowoff = gstart - r0;                      // 1 at top edge, else 0

  // Zero the halo rows not covered by the DMA (zero padding of the conv).
  if (cy == 0)         { for (int i = tid; i < BANDSTRIDE; i += 256) band[i] = 0.0f; }
  if (cy == CROWS - 1) { for (int i = tid; i < BANDSTRIDE; i += 256) band[9 * BANDSTRIDE + i] = 0.0f; }

#if HOG_TDM
  // ---- Tensor Data Mover: stage nrows x 512 f32 band into LDS, chunk-padded ----
  if (tid < 32) {   // single wave issues the descriptor (EXEC ignored by TDM)
    const unsigned lds_base = (unsigned)(uintptr_t)(&band[rowoff * BANDSTRIDE]);
    const unsigned long long ga =
        (unsigned long long)(uintptr_t)(x + ((size_t)img * IMG + gstart) * IMG);
    u32x4 g0;
    g0[0] = 1u;                                             // count=1, user descriptor
    g0[1] = lds_base;                                       // LDS byte address
    g0[2] = (unsigned)(ga & 0xFFFFFFFFu);                   // global addr [31:0]
    g0[3] = (unsigned)((ga >> 32) & 0x1FFFFFFu) | (2u << 30); // addr[56:32] | type=2
    i32x8 g1;
    // data_size=4B, pad_enable, pad_interval=16 dwords, pad_amount=4 dwords
    g1[0] = (int)((2u << 16) | (1u << 20) | (3u << 22) | (3u << 25));
    g1[1] = (int)((unsigned)(IMG & 0xFFFF) << 16);          // tensor_dim0 = 512 (lo16)
    g1[2] = (int)((unsigned)(IMG & 0xFFFF) << 16);          // dim0 hi=0 | tensor_dim1=512 lo16
    g1[3] = (int)((unsigned)(IMG & 0xFFFF) << 16);          // dim1 hi=0 | tile_dim0=512
    g1[4] = (int)(nrows & 0xFFFF);                          // tile_dim1 | tile_dim2=0
    g1[5] = IMG;                                            // tensor_dim0_stride = 512
    g1[6] = 0;
    g1[7] = 0;
    i32x4 gz = {0, 0, 0, 0};
#if __clang_major__ >= 23
    i32x8 gz8 = {0, 0, 0, 0, 0, 0, 0, 0};
    __builtin_amdgcn_tensor_load_to_lds(g0, g1, gz, gz, gz8, 0);
#else
    __builtin_amdgcn_tensor_load_to_lds(g0, g1, gz, gz, 0);
#endif
    __builtin_amdgcn_s_wait_tensorcnt(0);
  }
#else
  // Fallback: cooperative copy into the same chunk-padded layout.
  {
    const float* src = x + ((size_t)img * IMG + gstart) * IMG;
    for (int i = tid; i < nrows * IMG; i += 256) {
      const int r = i >> 9;
      const int c = i & (IMG - 1);
      band[(rowoff + r) * BANDSTRIDE + (c >> 4) * CHUNK + (c & 15)] = src[i];
    }
  }
#endif
  __syncthreads();

  // ---- Stencil: wave w -> pixel row 8*cy+w (band rows w, w+1, w+2) ----
  float v[3][18];   // columns 16L-1 .. 16L+16 for the three stencil rows
#pragma unroll
  for (int dr = 0; dr < 3; ++dr) {
    const int rb = (w + dr) * BANDSTRIDE + L * CHUNK;
    const float4 q0 = *(const float4*)(&band[rb + 0]);
    const float4 q1 = *(const float4*)(&band[rb + 4]);
    const float4 q2 = *(const float4*)(&band[rb + 8]);
    const float4 q3 = *(const float4*)(&band[rb + 12]);
    v[dr][1]  = q0.x; v[dr][2]  = q0.y; v[dr][3]  = q0.z; v[dr][4]  = q0.w;
    v[dr][5]  = q1.x; v[dr][6]  = q1.y; v[dr][7]  = q1.z; v[dr][8]  = q1.w;
    v[dr][9]  = q2.x; v[dr][10] = q2.y; v[dr][11] = q2.z; v[dr][12] = q2.w;
    v[dr][13] = q3.x; v[dr][14] = q3.y; v[dr][15] = q3.z; v[dr][16] = q3.w;
    v[dr][0]  = (L == 0)  ? 0.0f : band[rb - 5];       // chunk L-1, pos 15
    v[dr][17] = (L == 31) ? 0.0f : band[rb + CHUNK];   // chunk L+1, pos 0
  }

  // Cumulative histogram: cum[j][k] = sum of mag where ang >= k*pi/8 (k=0..7),
  // cum[j][8] = mass at ang == pi exactly. Indicators are monotone in k, so the
  // per-bin one-hot select collapses to a single telescoped difference per lane.
  float cum[2][NBINS];
#pragma unroll
  for (int j = 0; j < 2; ++j)
#pragma unroll
    for (int k = 0; k < NBINS; ++k) cum[j][k] = 0.0f;

  const float CK[7] = { 0.92387953f,  0.70710678f,  0.38268343f, 0.0f,
                       -0.38268343f, -0.70710678f, -0.92387953f};
  const float SK[7] = { 0.38268343f,  0.70710678f,  0.92387953f, 1.0f,
                        0.92387953f,  0.70710678f,  0.38268343f};

#pragma unroll
  for (int i = 0; i < 16; ++i) {
    const float a0 = v[0][i], a1 = v[0][i + 1], a2 = v[0][i + 2];
    const float m0 = v[1][i],                   m2 = v[1][i + 2];
    const float b0 = v[2][i], b1 = v[2][i + 1], b2 = v[2][i + 2];
    const float gx = (a2 - a0) + 2.0f * (m2 - m0) + (b2 - b0);
    const float gy = (b0 + 2.0f * b1 + b2) - (a0 + 2.0f * a1 + a2);
    const float mag = sqrtf(gx * gx + gy * gy);
    const float ax  = fabsf(gx);
    const int j = i >> 3;                             // cell 2L or 2L+1
    cum[j][0] += mag;                                 // ang >= 0 always
#pragma unroll
    for (int k = 1; k <= 7; ++k)
      cum[j][k] += (ax * CK[k - 1] - gy * SK[k - 1] >= 0.0f) ? mag : 0.0f;
    cum[j][8] += (ax == 0.0f && gy < 0.0f) ? mag : 0.0f;   // ang == pi
  }

  // ---- Per-wave partials: telescope cum -> per-bin, part[w][cell*9+bin] ----
#pragma unroll
  for (int j = 0; j < 2; ++j) {
#pragma unroll
    for (int b = 0; b < NBINS - 1; ++b)
      part[w * NCOLS + L * 18 + j * NBINS + b] = cum[j][b] - cum[j][b + 1];
    part[w * NCOLS + L * 18 + j * NBINS + 8] = cum[j][8];
  }
  __syncthreads();

  // ---- 8-row column-sum on the matrix pipe: D = ones(16x32) x B ----
  // Each partial is split into f16 hi/lo halves (K-slots r and 8+r), so the
  // f32 WMMA accumulator reconstructs the sum to ~f32 accuracy.
  v16h am;
#pragma unroll
  for (int i = 0; i < 16; ++i) am[i] = (_Float16)1.0f;

  for (int t = w; t < NCOLS / 16; t += 8) {   // 36 tiles over 8 waves (wave-uniform)
    v16h bm = {};
    if (L < 16) {
      const int col = t * 16 + L;
#pragma unroll
      for (int r = 0; r < 8; ++r) {
        const float m = part[r * NCOLS + col];
        const _Float16 hi = (_Float16)m;
        bm[r]     = hi;
        bm[8 + r] = (_Float16)(m - (float)hi);
      }
    }
    v8f c = {};
    c = __builtin_amdgcn_wmma_f32_16x16x32_f16(false, am, false, bm,
                                               (short)0, c, false, false);
    if (L < 16) {
      const int col = t * 16 + L;
      const int cx  = col / NBINS;
      const int b   = col - cx * NBINS;
      out[(((size_t)img * NBINS + b) * CROWS + cy) * CROWS + cx] = c[0];
    }
  }
}

extern "C" void kernel_launch(void* const* d_in, const int* in_sizes, int n_in,
                              void* d_out, int out_size, void* d_ws, size_t ws_size,
                              hipStream_t stream) {
  const float* x = (const float*)d_in[0];
  float* out = (float*)d_out;
  const int nimg = in_sizes[0] / (IMG * IMG);   // 64
  dim3 grid(CROWS, nimg);
  hog_cdna5_kernel<<<grid, 256, 0, stream>>>(x, out);
}
